// DiffusionPolicyHead_35261681500815
// MI455X (gfx1250) — compile-verified, matrix operand order checked by reading, not run on hardware
//
#include <hip/hip_runtime.h>
#include <cmath>

// Problem constants (match reference)
#define BATCH  32768
#define SDIM   128
#define ADIM   32
#define HID    256
#define NSTEPS 50

// Tiling
#define TB   64            // batch rows per block
#define STR  264           // padded LDS row stride (halfwords), 528B = 33*16B -> conflict-free b128
#define NBLK (BATCH / TB)  // 512 blocks
#define NTHR 512           // 16 wave32 per block

// Packed-weight tile table (each tile = 512 bf16 in WMMA-B register order)
#define T_W0A  0    // 16 tiles : ntile              (action part of W0, K=32)
#define T_W0S  16   // 64 tiles : ntile*4 + kt       (state  part of W0, K=128)
#define T_WH   80   // 384 tiles: layer*128 + ntile*8 + kt
#define T_WOUT 464  // 16 tiles : ntile*8 + kt       (Wout, N=32)
#define T_TOTAL 480

typedef __attribute__((ext_vector_type(16))) __bf16 v16bf;
typedef __attribute__((ext_vector_type(8)))  float  v8f;
typedef __attribute__((ext_vector_type(8)))  int    v8i;
typedef __attribute__((ext_vector_type(4)))  int    v4i;

struct StepConsts  { float c_in[NSTEPS]; float ca[NSTEPS]; float cb[NSTEPS]; float sigma0; };
struct NoiseConsts { float c_noise[NSTEPS]; };

__device__ __forceinline__ unsigned short f2bf(float f) {
  unsigned int u = __builtin_bit_cast(unsigned int, f);
  u += 0x7FFFu + ((u >> 16) & 1u);            // round-to-nearest-even
  return (unsigned short)(u >> 16);
}

__device__ __forceinline__ v8f zero8() {
  v8f r;
#pragma unroll
  for (int i = 0; i < 8; i++) r[i] = 0.f;
  return r;
}

// K index held by (lane, halfword i) for 16-bit A/B operands (ISA 7.12.2 layout):
// lanes 0-15 hold K {0..7,16..23}, lanes 16-31 hold K {8..15,24..31}.
__device__ __forceinline__ int kmap(int i, int laneHi) {
  return (i < 8 ? i : i + 8) + (laneHi ? 8 : 0);
}

// A fragment from row-major bf16 LDS tile: two contiguous 16B chunks per lane.
__device__ __forceinline__ v16bf load_afrag(const unsigned short* buf, int mtile, int kt,
                                            int ln, int hl) {
  const unsigned short* p = buf + (mtile * 16 + ln) * STR + kt * 32 + hl * 8;
  union { v4i h[2]; v16bf f; } u;
  u.h[0] = *(const v4i*)p;
  u.h[1] = *(const v4i*)(p + 16);
  return u.f;
}

// B fragment: pre-packed in register order -> one contiguous 32B load per lane (L2-resident).
__device__ __forceinline__ v16bf load_bfrag(const unsigned short* wp, int tile, int lane) {
  union { v8i w; v16bf f; } u;
  u.w = *(const v8i*)(wp + (size_t)tile * 512 + lane * 16);
  return u.f;
}

// relu + bf16 + scatter C tile (f32 16x16 layout: lane%16=N, vgpr v -> M=v(+8 for hi lanes))
__device__ __forceinline__ void store_h(unsigned short* dst, v8f c, int mtile, int ntile,
                                        int ln, int hl) {
#pragma unroll
  for (int v = 0; v < 8; v++) {
    int r = mtile * 16 + v + hl * 8;
    float x = c[v];
    x = x > 0.f ? x : 0.f;
    dst[r * STR + ntile * 16 + ln] = f2bf(x);
  }
}

// ---------------- prep kernel 1: pack all weights to bf16 in WMMA-B register order ----
__global__ void pack_weights(const float* __restrict__ W0, const float* __restrict__ Wh,
                             const float* __restrict__ Wout, unsigned short* __restrict__ wp) {
  int t = blockIdx.x;        // tile id
  int j = threadIdx.x;       // element in tile, 0..511
  int lane = j >> 4, i = j & 15;
  int nl = lane & 15;
  int kl = kmap(i, lane & 16);
  float v;
  if (t < 16) {                                   // W0 action part rows 0..31
    v = W0[kl * HID + t * 16 + nl];
  } else if (t < 80) {                            // W0 state part rows 96..223
    int u = t - 16, ntile = u >> 2, kt = u & 3;
    v = W0[(96 + kt * 32 + kl) * HID + ntile * 16 + nl];
  } else if (t < 464) {                           // Wh[3][256][256]
    int u = t - 80, layer = u >> 7, w = u & 127;
    int ntile = w >> 3, kt = w & 7;
    v = Wh[layer * HID * HID + (kt * 32 + kl) * HID + ntile * 16 + nl];
  } else {                                        // Wout[256][32]
    int u = t - 464, ntile = u >> 3, kt = u & 7;
    v = Wout[(kt * 32 + kl) * ADIM + ntile * 16 + nl];
  }
  wp[(size_t)t * 512 + j] = f2bf(v);
}

// ---------------- prep kernel 2: per-step effective bias b0 + emb(sigma) @ W0[32:96] ----
__global__ void build_b0eff(const float* __restrict__ W0, const float* __restrict__ b0,
                            float* __restrict__ beff, NoiseConsts nc) {
  int step = blockIdx.x, c = threadIdx.x;
  float cn = nc.c_noise[step];
  float acc = b0[c];
#pragma unroll 8
  for (int j = 0; j < 32; j++) {
    float f = __expf(-9.2103403719761836f * (float)j / 32.0f);   // ln(10000)
    float ang = cn * f;
    acc += __sinf(ang) * W0[(32 + j) * HID + c] + __cosf(ang) * W0[(64 + j) * HID + c];
  }
  beff[step * HID + c] = acc;
}

// ---------------- main kernel: whole 50-step diffusion chain per batch tile ------------
__global__ __launch_bounds__(NTHR) void diffusion_head(
    const float* __restrict__ state, const float* __restrict__ noise,
    const float* __restrict__ bh, const float* __restrict__ bout,
    const unsigned short* __restrict__ wp, const float* __restrict__ beff,
    float* __restrict__ out, StepConsts sc) {
  __shared__ __align__(16) unsigned short buf[TB * STR];  // 33.8 KB activation tile (bf16)
  __shared__ float act[TB * ADIM];                        // 8 KB action state (f32)

  const int tid  = threadIdx.x;
  const int lane = tid & 31, wv = tid >> 5;               // 16 waves; wv = hidden N-tile
  const int ln   = lane & 15, hl = (lane >> 4) & 1;
  const size_t row0 = (size_t)blockIdx.x * TB;

  // action0 = sigma0 * init_noise
  for (int e = tid; e < TB * ADIM; e += NTHR)
    act[e] = sc.sigma0 * noise[row0 * ADIM + e];
  // state tile -> bf16 LDS cols 0..127
  for (int e = tid; e < TB * SDIM; e += NTHR) {
    int r = e >> 7, c = e & 127;
    buf[r * STR + c] = f2bf(state[(row0 + r) * SDIM + c]);
  }
  __syncthreads();

  // s0 = state @ W0_state, kept in accumulator VGPRs for all 50 steps
  v8f s0[4];
#pragma unroll
  for (int m = 0; m < 4; m++) s0[m] = zero8();
#pragma unroll
  for (int kt = 0; kt < 4; kt++) {
    v16bf b = load_bfrag(wp, T_W0S + wv * 4 + kt, lane);
#pragma unroll
    for (int m = 0; m < 4; m++) {
      v16bf a = load_afrag(buf, m, kt, ln, hl);
      s0[m] = __builtin_amdgcn_wmma_f32_16x16x32_bf16(false, a, false, b, (short)0, s0[m],
                                                      false, false);
    }
  }
  __syncthreads();

#pragma unroll 1
  for (int step = 0; step < NSTEPS; step++) {
    // scaled action -> buf cols 0..31 (bf16)
    float cin = sc.c_in[step];
    for (int e = tid; e < TB * ADIM; e += NTHR) {
      int r = e >> 5, c = e & 31;
      buf[r * STR + c] = f2bf(cin * act[e]);
    }
    __syncthreads();

    // layer 0: h0 = relu(s0 + b0eff + act_bf @ W0_act)   (K=32, one k-tile)
    {
      float bias = beff[step * HID + wv * 16 + ln];
      v8f c4[4];
#pragma unroll
      for (int m = 0; m < 4; m++) c4[m] = s0[m] + bias;
      v16bf b = load_bfrag(wp, T_W0A + wv, lane);
#pragma unroll
      for (int m = 0; m < 4; m++) {
        v16bf a = load_afrag(buf, m, 0, ln, hl);
        c4[m] = __builtin_amdgcn_wmma_f32_16x16x32_bf16(false, a, false, b, (short)0, c4[m],
                                                        false, false);
      }
      __syncthreads();  // read phase done -> safe to overwrite in place
#pragma unroll
      for (int m = 0; m < 4; m++) store_h(buf, c4[m], m, wv, ln, hl);
      __syncthreads();
    }

    // 3 hidden layers, in-place (read all -> barrier -> write all)
#pragma unroll 1
    for (int g = 0; g < 3; g++) {
      v8f c4[4];
#pragma unroll
      for (int m = 0; m < 4; m++) c4[m] = zero8();
#pragma unroll
      for (int kt = 0; kt < 8; kt++) {
        v16bf b = load_bfrag(wp, T_WH + g * 128 + wv * 8 + kt, lane);
#pragma unroll
        for (int m = 0; m < 4; m++) {
          v16bf a = load_afrag(buf, m, kt, ln, hl);
          c4[m] = __builtin_amdgcn_wmma_f32_16x16x32_bf16(false, a, false, b, (short)0, c4[m],
                                                          false, false);
        }
      }
      float bias = bh[g * HID + wv * 16 + ln];
      __syncthreads();
#pragma unroll
      for (int m = 0; m < 4; m++) store_h(buf, c4[m] + bias, m, wv, ln, hl);
      __syncthreads();
    }

    // output layer (N=32 -> 8 waves) + fused DDIM update of action in LDS
    if (wv < 8) {
      int mt = wv >> 1, nt = wv & 1;
      v8f c = zero8();
#pragma unroll
      for (int kt = 0; kt < 8; kt++) {
        v16bf b = load_bfrag(wp, T_WOUT + nt * 8 + kt, lane);
        v16bf a = load_afrag(buf, mt, kt, ln, hl);
        c = __builtin_amdgcn_wmma_f32_16x16x32_bf16(false, a, false, b, (short)0, c,
                                                    false, false);
      }
      float bo  = bout[nt * 16 + ln];
      float caS = sc.ca[step], cbS = sc.cb[step];
#pragma unroll
      for (int v = 0; v < 8; v++) {
        int r  = mt * 16 + v + hl * 8;
        int ci = r * ADIM + nt * 16 + ln;
        float ao = act[ci];
        act[ci] = caS * ao + cbS * (c[v] + bo);   // a' = (r+(1-r)c_skip)a + (1-r)c_out*inner
      }
    }
    __syncthreads();
  }

  for (int e = tid; e < TB * ADIM; e += NTHR)
    out[row0 * ADIM + e] = act[e];
}

// ---------------- host ------------------------------------------------------------------
extern "C" void kernel_launch(void* const* d_in, const int* in_sizes, int n_in,
                              void* d_out, int out_size, void* d_ws, size_t ws_size,
                              hipStream_t stream) {
  const float* state = (const float*)d_in[0];
  const float* noise = (const float*)d_in[1];
  const float* W0    = (const float*)d_in[2];
  const float* b0    = (const float*)d_in[3];
  const float* Wh    = (const float*)d_in[4];
  const float* bhp   = (const float*)d_in[5];
  const float* Wout  = (const float*)d_in[6];
  const float* boutp = (const float*)d_in[7];
  float* outp = (float*)d_out;

  unsigned short* wp = (unsigned short*)d_ws;
  float* beff = (float*)((char*)d_ws + (size_t)T_TOTAL * 512 * sizeof(unsigned short));

  // Karras sigma schedule + EDM/DDIM per-step scalars (host math, graph-capture safe)
  double sig[NSTEPS + 1];
  double min_r = pow(0.001, 1.0 / 7.0), max_r = pow(80.0, 1.0 / 7.0);
  for (int i = 0; i < NSTEPS; i++) {
    double ramp = (double)i / (double)(NSTEPS - 1);
    sig[i] = pow(max_r + ramp * (min_r - max_r), 7.0);
  }
  sig[NSTEPS] = 0.0;

  StepConsts sc; NoiseConsts nc;
  for (int i = 0; i < NSTEPS; i++) {
    double s = sig[i], sn = sig[i + 1];
    double var = s * s + 1.0;
    double cin   = 1.0 / sqrt(var);
    double cskip = 1.0 / var;
    double cout  = s / sqrt(var);
    double ratio = sn / s;
    sc.c_in[i] = (float)cin;
    sc.ca[i]   = (float)(ratio + (1.0 - ratio) * cskip);
    sc.cb[i]   = (float)((1.0 - ratio) * cout);
    nc.c_noise[i] = (float)log(s);
  }
  sc.sigma0 = (float)sig[0];

  pack_weights<<<T_TOTAL, 512, 0, stream>>>(W0, Wh, Wout, wp);
  build_b0eff<<<NSTEPS, 256, 0, stream>>>(W0, b0, beff, nc);
  diffusion_head<<<NBLK, NTHR, 0, stream>>>(state, noise, bhp, boutp, wp, beff, outp, sc);
}
